// MHCABlock_86552180949230
// MI455X (gfx1250) — matrix operand order repaired and checked by reading, not compiled, not run on hardware
//
#include <hip/hip_runtime.h>
#include <hip/hip_bf16.h>
#include <math.h>

// ---------------------------------------------------------------------------
// MHCA block for MI455X (gfx1250, wave32, WMMA f32_16x16x32_f16,
// async global->LDS staging for GEMM A-tiles)
// ---------------------------------------------------------------------------

typedef __attribute__((ext_vector_type(8)))  _Float16 v8h;
typedef __attribute__((ext_vector_type(16))) _Float16 v16h;
typedef __attribute__((ext_vector_type(8)))  float    v8f;
typedef __attribute__((ext_vector_type(4)))  int      v4i;

union F16x16 { v16h v; v8h h[2]; _Float16 e[16]; };

__device__ __forceinline__ v8f wmma_f16(v16h a, v16h b, v8f c) {
  // (neg_a, A, neg_b, B, c_mod, C, reuse_a, reuse_b)
  return __builtin_amdgcn_wmma_f32_16x16x32_f16(false, a, false, b, (short)0, c,
                                                false, false);
}

static __device__ __forceinline__ v8f vzero8() {
  v8f z = {0.f, 0.f, 0.f, 0.f, 0.f, 0.f, 0.f, 0.f};
  return z;
}

// ---- async global -> LDS copy of 16 bytes per lane (ASYNCcnt tracked) ------
__device__ __forceinline__ void async_copy16(const void* gptr, void* lptr) {
#if __has_builtin(__builtin_amdgcn_global_load_async_to_lds_b128)
  __builtin_amdgcn_global_load_async_to_lds_b128(
      (__attribute__((address_space(1))) v4i*)(void*)gptr,
      (__attribute__((address_space(3))) v4i*)lptr, 0, 0);
#else
  unsigned lds_off =
      (unsigned)(uintptr_t)(__attribute__((address_space(3))) void*)lptr;
  asm volatile("global_load_async_to_lds_b128 %0, %1, off"
               :
               : "v"(lds_off), "v"(gptr)
               : "memory");
#endif
}

__device__ __forceinline__ void async_wait0() {
#if __has_builtin(__builtin_amdgcn_s_wait_asynccnt)
  __builtin_amdgcn_s_wait_asynccnt(0);
#else
  asm volatile("s_wait_asynccnt 0" ::: "memory");
#endif
}

// ---------------- fp32 -> f16 conversion (grid-stride) ----------------------
__global__ __launch_bounds__(256)
void cvt_f32_to_f16(const float* __restrict__ in, _Float16* __restrict__ out, int n) {
  int i = blockIdx.x * 256 + threadIdx.x;
  int stride = gridDim.x * 256;
  for (; i < n; i += stride) out[i] = (_Float16)in[i];
}

// ---------------- epilogue functors -----------------------------------------
struct EpiHeads {            // scatter (m = b*1024+tok, n = h*64+d) -> [B,H,N,D]
  _Float16* out;
  __device__ void operator()(int m, int n, float v) const {
    int b = m >> 10, tok = m & 1023, h = n >> 6, d = n & 63;
    out[(((size_t)(b * 12 + h)) * 1024 + tok) * 64 + d] = (_Float16)v;
  }
};
struct EpiResid {            // out = resid + v   (residual 1, fp32, ld = 768)
  float* out; const float* resid;
  __device__ void operator()(int m, int n, float v) const {
    size_t i = (size_t)m * 768 + n;
    out[i] = resid[i] + v;
  }
};
struct EpiGelu {             // exact GELU, store f16
  _Float16* out; int ld;
  __device__ void operator()(int m, int n, float v) const {
    float g = 0.5f * v * (1.0f + erff(v * 0.70710678118654752f));
    out[(size_t)m * ld + n] = (_Float16)g;
  }
};
struct EpiAddOut {           // out += v          (residual 2, fp32, ld = 768)
  float* out;
  __device__ void operator()(int m, int n, float v) const {
    out[(size_t)m * 768 + n] += v;
  }
};

// ---------------- generic Linear GEMM ---------------------------------------
// C[m,n] = sum_k A[m,k] * W[n,k] (+ bias[n]);  A: MxK f16, W: NxK f16.
// Block: 256 threads = 8 waves in 2(M)x4(N); wave tile 32x64; block tile 64x256.
// A block-tile (64 rows x 32 K) is staged in LDS via async global->LDS DMA,
// double-buffered so the copy of step s+1 overlaps the WMMAs of step s.
// B fragments (contiguous rows of W) are read direct from global (L2-resident).
// Grid: (M/64, N/256). All K multiples of 32; rows 16B aligned.
template <class Epi>
__global__ __launch_bounds__(256)
void gemm_f16(const _Float16* __restrict__ A, const _Float16* __restrict__ W,
              const float* __restrict__ bias, int K, Epi epi) {
  __shared__ __align__(16) char smem[2 * 4096];   // 2 buffers x (64 rows x 64B)

  const int tid   = threadIdx.x;
  const int lane  = tid & 31;
  const int wave  = tid >> 5;
  const int wm    = wave & 1, wn = wave >> 1;
  const int lhalf = lane >> 4, lmod = lane & 15;
  const int rowBase = blockIdx.x * 64 + wm * 32;
  const int colBase = blockIdx.y * 256 + wn * 64;

  // copy mapping: thread -> (row, 16B segment) of the 64x32-half A tile
  const int cpRow = tid >> 2;
  const int cpSeg = (tid & 3) * 16;
  const char* Abase =
      (const char*)(A + (size_t)(blockIdx.x * 64 + cpRow) * K) + cpSeg;

  v8f acc[2][4];
#pragma unroll
  for (int i = 0; i < 2; i++)
#pragma unroll
    for (int j = 0; j < 4; j++) acc[i][j] = vzero8();

  // prefetch k-step 0 into buffer 0
  async_copy16(Abase, smem + cpRow * 64 + cpSeg);

  const int steps = K >> 5;
  for (int s = 0; s < steps; s++) {
    async_wait0();        // my async stores to LDS landed
    __syncthreads();      // everyone's stores landed; prev buffer fully consumed

    if (s + 1 < steps) {  // overlap next DMA with this step's WMMAs
      async_copy16(Abase + (size_t)(s + 1) * 64,
                   smem + ((s + 1) & 1) * 4096 + cpRow * 64 + cpSeg);
    }

    const int kk = s << 5;
    const char* abuf = smem + (s & 1) * 4096;

    F16x16 a[2], b[4];
#pragma unroll
    for (int mi = 0; mi < 2; mi++) {
      // 16-bit A-frag: lanes 0-15 hold K 0-7 & 16-23, lanes 16-31 hold 8-15 & 24-31
      const char* ap = abuf + (wm * 32 + mi * 16 + lmod) * 64 + lhalf * 16;
      a[mi].h[0] = *(const v8h*)ap;          // ds_load_b128
      a[mi].h[1] = *(const v8h*)(ap + 32);
    }
#pragma unroll
    for (int ni = 0; ni < 4; ni++) {
      // B-frag column (= row of W): lanes 0-15 K 0-7 & 16-23, lanes 16-31 K 8-15 & 24-31
      const _Float16* wp =
          W + (size_t)(colBase + ni * 16 + lmod) * K + kk + lhalf * 8;
      b[ni].h[0] = *(const v8h*)wp;
      b[ni].h[1] = *(const v8h*)(wp + 16);
    }
#pragma unroll
    for (int mi = 0; mi < 2; mi++)
#pragma unroll
      for (int ni = 0; ni < 4; ni++)
        acc[mi][ni] = wmma_f16(a[mi].v, b[ni].v, acc[mi][ni]);
  }

#pragma unroll
  for (int mi = 0; mi < 2; mi++)
#pragma unroll
    for (int ni = 0; ni < 4; ni++) {
#pragma unroll
      for (int j = 0; j < 8; j++) {
        int m = rowBase + mi * 16 + j + lhalf * 8;   // D layout: VGPR j -> M=j (+8 hi half)
        int n = colBase + ni * 16 + lmod;
        float v = acc[mi][ni][j] + (bias ? bias[n] : 0.0f);
        epi(m, n, v);
      }
    }
}

// ---------------- softmax over token axis (per (b,h,d) column) --------------
// K stored [B,H,N,D]; one wave per column, in-place.
__global__ __launch_bounds__(256)
void col_softmax_kernel(_Float16* Kh) {
  const int col  = blockIdx.x * 8 + (threadIdx.x >> 5);   // 0 .. B*H*D-1 (12288)
  const int lane = threadIdx.x & 31;
  const int bh = col >> 6, d = col & 63;
  _Float16* base = Kh + ((size_t)bh * 1024) * 64 + d;

  float vals[32];
  float mx = -1e30f;
#pragma unroll
  for (int i = 0; i < 32; i++) {
    float t = (float)base[(size_t)(i * 32 + lane) * 64];
    vals[i] = t;
    mx = fmaxf(mx, t);
  }
#pragma unroll
  for (int off = 16; off > 0; off >>= 1) mx = fmaxf(mx, __shfl_xor(mx, off, 32));
  float s = 0.0f;
#pragma unroll
  for (int i = 0; i < 32; i++) { vals[i] = expf(vals[i] - mx); s += vals[i]; }
#pragma unroll
  for (int off = 16; off > 0; off >>= 1) s += __shfl_xor(s, off, 32);
  float inv = 1.0f / s;
#pragma unroll
  for (int i = 0; i < 32; i++)
    base[(size_t)(i * 32 + lane) * 64] = (_Float16)(vals[i] * inv);
}

// ---------------- kv = k_softmax^T @ v, stored transposed -------------------
// Per (b,h): kvT[d2*64+d1] = sum_n ks[n,d1]*v[n,d2]. 4 waves, wave w: d1 in [16w,16w+16).
__global__ __launch_bounds__(128)
void kv_kernel(const _Float16* __restrict__ KS, const _Float16* __restrict__ V,
               _Float16* __restrict__ kvT) {
  const int bh   = blockIdx.x;
  const int lane = threadIdx.x & 31;
  const int wave = threadIdx.x >> 5;
  const int lhalf = lane >> 4, lmod = lane & 15;
  const _Float16* ks = KS + (size_t)bh * 1024 * 64;
  const _Float16* v  = V  + (size_t)bh * 1024 * 64;

  v8f acc[4];
#pragma unroll
  for (int j = 0; j < 4; j++) acc[j] = vzero8();

  const int m = wave * 16 + lmod;                       // d1 row of A = ks^T
  for (int kk = 0; kk < 1024; kk += 32) {
    F16x16 a, b[4];
#pragma unroll
    for (int i = 0; i < 8; i++) {                       // strided gather of ks column
      int n0 = kk + lhalf * 16 + i;
      a.e[i]     = ks[(size_t)n0 * 64 + m];
      a.e[i + 8] = ks[(size_t)(n0 + 8) * 64 + m];
    }
#pragma unroll
    for (int ni = 0; ni < 4; ni++) {
      int c = ni * 16 + lmod;                           // d2 column
#pragma unroll
      for (int i = 0; i < 16; i++)
        b[ni].e[i] = v[(size_t)(kk + lhalf * 16 + i) * 64 + c];
    }
#pragma unroll
    for (int ni = 0; ni < 4; ni++) acc[ni] = wmma_f16(a.v, b[ni].v, acc[ni]);
  }

  _Float16* o = kvT + (size_t)bh * 4096;
#pragma unroll
  for (int ni = 0; ni < 4; ni++)
#pragma unroll
    for (int j = 0; j < 8; j++) {
      int d1 = wave * 16 + j + lhalf * 8;
      int d2 = ni * 16 + lmod;
      o[(size_t)d2 * 64 + d1] = (_Float16)acc[ni][j];   // store transposed
    }
}

// ---------------- factor_att = q @ kv, scatter to [B,N,C] -------------------
// grid (1024/256, B*H); per (b,h): C[tok,d2] = sum_d1 q[tok,d1]*kvT[d2,d1]
__global__ __launch_bounds__(256)
void factor_att_kernel(const _Float16* __restrict__ Q, const _Float16* __restrict__ kvT,
                       _Float16* __restrict__ att) {
  const int bh   = blockIdx.y;
  const int lane = threadIdx.x & 31;
  const int wave = threadIdx.x >> 5;
  const int lhalf = lane >> 4, lmod = lane & 15;
  const int rowBase = blockIdx.x * 256 + wave * 32;
  const _Float16* Ab = Q   + (size_t)bh * 1024 * 64;
  const _Float16* Wb = kvT + (size_t)bh * 64 * 64;

  v8f acc[2][4];
#pragma unroll
  for (int i = 0; i < 2; i++)
#pragma unroll
    for (int j = 0; j < 4; j++) acc[i][j] = vzero8();

#pragma unroll
  for (int kk = 0; kk < 64; kk += 32) {
    F16x16 a[2], b[4];
#pragma unroll
    for (int mi = 0; mi < 2; mi++) {
      const _Float16* ap = Ab + (size_t)(rowBase + mi * 16 + lmod) * 64 + kk + lhalf * 8;
      a[mi].h[0] = *(const v8h*)ap;
      a[mi].h[1] = *(const v8h*)(ap + 16);
    }
#pragma unroll
    for (int ni = 0; ni < 4; ni++) {
      const _Float16* wp = Wb + (size_t)(ni * 16 + lmod) * 64 + kk + lhalf * 8;
      b[ni].h[0] = *(const v8h*)wp;
      b[ni].h[1] = *(const v8h*)(wp + 16);
    }
#pragma unroll
    for (int mi = 0; mi < 2; mi++)
#pragma unroll
      for (int ni = 0; ni < 4; ni++)
        acc[mi][ni] = wmma_f16(a[mi].v, b[ni].v, acc[mi][ni]);
  }

  const int b_ = bh / 12, h_ = bh % 12;
#pragma unroll
  for (int mi = 0; mi < 2; mi++)
#pragma unroll
    for (int ni = 0; ni < 4; ni++)
#pragma unroll
      for (int j = 0; j < 8; j++) {
        int tok = rowBase + mi * 16 + j + lhalf * 8;
        int n = ni * 16 + lmod;
        att[((size_t)b_ * 1024 + tok) * 768 + h_ * 64 + n] = (_Float16)acc[mi][ni][j];
      }
}

// ---------------- LayerNorm (eps=1e-6) over fp32 rows -> f16 ----------------
__global__ __launch_bounds__(256)
void ln_kernel(const float* __restrict__ X, const float* __restrict__ gamma,
               const float* __restrict__ beta, _Float16* __restrict__ cur) {
  const int row = blockIdx.x;
  const int tid = threadIdx.x;
  const float* xr = X + (size_t)row * 768;
  float v0 = xr[tid], v1 = xr[tid + 256], v2 = xr[tid + 512];

  __shared__ float red[256];
  red[tid] = v0 + v1 + v2;
  __syncthreads();
  for (int off = 128; off > 0; off >>= 1) {
    if (tid < off) red[tid] += red[tid + off];
    __syncthreads();
  }
  float mu = red[0] * (1.0f / 768.0f);
  __syncthreads();

  float d0 = v0 - mu, d1 = v1 - mu, d2 = v2 - mu;
  red[tid] = d0 * d0 + d1 * d1 + d2 * d2;
  __syncthreads();
  for (int off = 128; off > 0; off >>= 1) {
    if (tid < off) red[tid] += red[tid + off];
    __syncthreads();
  }
  float rstd = rsqrtf(red[0] * (1.0f / 768.0f) + 1e-6f);

  _Float16* o = cur + (size_t)row * 768;
  o[tid]       = (_Float16)(d0 * rstd * gamma[tid]       + beta[tid]);
  o[tid + 256] = (_Float16)(d1 * rstd * gamma[tid + 256] + beta[tid + 256]);
  o[tid + 512] = (_Float16)(d2 * rstd * gamma[tid + 512] + beta[tid + 512]);
}

// ---------------------------------------------------------------------------
extern "C" void kernel_launch(void* const* d_in, const int* in_sizes, int n_in,
                              void* d_out, int out_size, void* d_ws, size_t ws_size,
                              hipStream_t stream) {
  (void)in_sizes; (void)n_in; (void)out_size; (void)ws_size;

  const float* x    = (const float*)d_in[0];
  const float* ctx  = (const float*)d_in[1];
  const float* Wq   = (const float*)d_in[2];
  const float* bq   = (const float*)d_in[3];
  const float* Wk   = (const float*)d_in[4];
  const float* bk   = (const float*)d_in[5];
  const float* Wv   = (const float*)d_in[6];
  const float* bv   = (const float*)d_in[7];
  const float* Wp   = (const float*)d_in[8];
  const float* bp   = (const float*)d_in[9];
  const float* W1   = (const float*)d_in[10];
  const float* b1   = (const float*)d_in[11];
  const float* W2   = (const float*)d_in[12];
  const float* b2   = (const float*)d_in[13];
  const float* gam  = (const float*)d_in[14];
  const float* bet  = (const float*)d_in[15];
  float* out = (float*)d_out;

  constexpr int M   = 16384;          // B*N
  constexpr int C   = 768;
  constexpr int HID = 2304;
  constexpr int SX  = M * C;          // activation-sized f16 buffer
  constexpr int W768 = C * C;
  constexpr int WH   = HID * C;
  constexpr int KVT  = 192 * 64 * 64; // B*H*D*D
  constexpr int SH   = M * HID;

  _Float16* base = (_Float16*)d_ws;
  size_t off = 0;
  auto alloc = [&](size_t n) {
    _Float16* p = base + off;
    off += (n + 127) & ~(size_t)127;
    return p;
  };
  _Float16* xb  = alloc(SX);   // x (f16); reused later as `cur`
  _Float16* cb  = alloc(SX);   // context (f16); reused later as `att`
  _Float16* qh  = alloc(SX);   // Q  [B,H,N,D]
  _Float16* kh  = alloc(SX);   // K  [B,H,N,D] -> softmax in place
  _Float16* vh  = alloc(SX);   // V  [B,H,N,D]
  _Float16* wqb = alloc(W768);
  _Float16* wkb = alloc(W768);
  _Float16* wvb = alloc(W768);
  _Float16* wpb = alloc(W768);
  _Float16* w1b = alloc(WH);
  _Float16* w2b = alloc(WH);
  _Float16* kvt = alloc(KVT);
  _Float16* hb  = alloc(SH);   // MLP hidden
  _Float16* att = cb;          // reuse
  _Float16* cur = xb;          // reuse

  auto cvt = [&](const float* src, _Float16* dst, int n) {
    int grid = (n + 255) / 256; if (grid > 4096) grid = 4096;
    cvt_f32_to_f16<<<dim3(grid), dim3(256), 0, stream>>>(src, dst, n);
  };
  cvt(x, xb, SX);
  cvt(ctx, cb, SX);
  cvt(Wq, wqb, W768);
  cvt(Wk, wkb, W768);
  cvt(Wv, wvb, W768);
  cvt(Wp, wpb, W768);
  cvt(W1, w1b, WH);
  cvt(W2, w2b, WH);

  const dim3 blk(256);
  const dim3 gQKV(M / 64, C / 256);        // (256, 3)

  // Q/K/V projections -> head-major f16
  gemm_f16<<<gQKV, blk, 0, stream>>>(xb, wqb, bq, C, EpiHeads{qh});
  gemm_f16<<<gQKV, blk, 0, stream>>>(cb, wkb, bk, C, EpiHeads{kh});
  gemm_f16<<<gQKV, blk, 0, stream>>>(cb, wvb, bv, C, EpiHeads{vh});

  // softmax over token axis, per (b,h,d) column, in place
  col_softmax_kernel<<<dim3(16 * 12 * 64 / 8), blk, 0, stream>>>(kh);

  // kv = ks^T @ v  (stored transposed per head)
  kv_kernel<<<dim3(192), dim3(128), 0, stream>>>(kh, vh, kvt);

  // factor_att = q @ kv -> att [B,N,C] f16
  factor_att_kernel<<<dim3(1024 / 256, 192), blk, 0, stream>>>(qh, kvt, att);

  // residual 1: out = x + att @ Wp^T + bp   (fp32, written to d_out)
  gemm_f16<<<gQKV, blk, 0, stream>>>(att, wpb, bp, C, EpiResid{out, x});

  // LayerNorm -> cur (f16)
  ln_kernel<<<dim3(M), blk, 0, stream>>>(out, gam, bet, cur);

  // MLP: h = GELU(cur @ W1^T + b1);  out += h @ W2^T + b2
  gemm_f16<<<dim3(M / 64, HID / 256), blk, 0, stream>>>(cur, w1b, b1, C, EpiGelu{hb, HID});
  gemm_f16<<<dim3(M / 64, C / 256), blk, 0, stream>>>(hb, w2b, b2, HID, EpiAddOut{out});
}
